// NequIPWrapper_58798102283031
// MI455X (gfx1250) — matrix operand order repaired and checked by reading, not compiled, not run on hardware
//
#include <hip/hip_runtime.h>
#include <hip/hip_bf16.h>

#define NATOMS 50000
#define NEDGES 800000
#define FDIM   32
#define NBAS   8
#define RCUT_F 20.0f
#define EV2KJ_F 96.4853f
#define NM2A_F 10.0f
#define FSCALE_F (EV2KJ_F * NM2A_F)
#define PI_F 3.14159265358979f

typedef __attribute__((ext_vector_type(16))) _Float16 v16h;
typedef __attribute__((ext_vector_type(8)))  float    v8f;

// ---- workspace layout (float offsets) ----
#define OFF_S   0
#define OFF_V   (NATOMS*FDIM)              // s: N*32
#define OFF_DS  (OFF_V + NATOMS*FDIM*3)    // v: N*96
#define OFF_DV  (OFF_DS + NATOMS*FDIM)     // ds: N*32
#define OFF_G   (OFF_DV + NATOMS*FDIM*3)   // dv: N*96
#define OFF_SC  (OFF_G + NATOMS*3)         // grad pos: N*3
#define WS_FLOATS (OFF_SC + 8)             // scalars: [0]=E_eV, [1..3]=net, [4]=msum

__device__ __forceinline__ float sigf(float x) { return 1.0f / (1.0f + __expf(-x)); }

__device__ __forceinline__ v8f wmma_f16(v16h a, v16h b, v8f c) {
  return __builtin_amdgcn_wmma_f32_16x16x32_f16(false, a, false, b, (short)0, c, false, false);
}

// A fragment: 16x32 f16 tile, rows = edges. buf is [row][k] with row-stride ld.
// Lanes 0-15: row=lane, K in {kofs+0..7, kofs+16..23}; lanes 16-31: same row, K+8 / K+24.
__device__ __forceinline__ v16h ldsA(const _Float16* buf, int ld, int mtile, int kofs, int lane) {
  int row = mtile * 16 + (lane & 15);
  int kb  = kofs + ((lane & 16) ? 8 : 0);
  union { v16h v; _Float16 h[16]; } u;
#pragma unroll
  for (int i = 0; i < 8; ++i) {
    int k = kb + ((i < 4) ? 2 * i : 16 + 2 * (i - 4));
    u.h[2 * i]     = buf[row * ld + k];
    u.h[2 * i + 1] = buf[row * ld + k + 1];
  }
  return u.v;
}

// B fragment: 32x16 f16 tile (KxN). bt is stored column-major-per-N: bt[n][k], k contiguous (stride ldk).
// Lanes 0-15 hold K in {kofs+0..7, kofs+16..23}; lanes 16-31: K+8 / K+24, col = ntile*16 + lane%16.
__device__ __forceinline__ v16h ldsB(const _Float16* bt, int ldk, int ntile, int kofs, int lane) {
  int col = ntile * 16 + (lane & 15);
  int kb  = kofs + ((lane & 16) ? 8 : 0);
  union { v16h v; _Float16 h[16]; } u;
#pragma unroll
  for (int i = 0; i < 16; ++i) u.h[i] = bt[col * ldk + kb + ((i < 8) ? i : 8 + i)];
  return u.v;
}

// ------------------------------------------------------------------ zero ws
__global__ void zero_ws_kernel(float* __restrict__ w, int n) {
  int i = blockIdx.x * 256 + threadIdx.x;
  if (i < n) w[i] = 0.0f;
}

// ------------------------------------------------------------------ edge forward
#define WPB_F 4
__global__ __launch_bounds__(128) void edge_fwd_kernel(
    const float* __restrict__ pos, const float* __restrict__ W1,
    const float* __restrict__ b1, const float* __restrict__ W2,
    const int* __restrict__ types, const int* __restrict__ ei,
    const float* __restrict__ temb, float* __restrict__ Sacc, float* __restrict__ Vacc)
{
  __shared__ _Float16 sBas[WPB_F][32 * 32];
  __shared__ _Float16 sAct[WPB_F][32 * 32];
  __shared__ float    sOut[WPB_F][32 * 64];
  __shared__ _Float16 sW1T[32 * 32];   // [n][k]: W1[k][n], zero-padded K 8->32
  __shared__ _Float16 sW2T[64 * 32];   // [c][f]: W2[f][c]
  __shared__ float    sB1[32];
  __shared__ float    sTE[96];

  int tid = threadIdx.x;
  for (int i = tid; i < 32 * 32; i += 128) { int n = i >> 5, k = i & 31;
    sW1T[n * 32 + k] = (_Float16)((k < NBAS) ? W1[k * FDIM + n] : 0.0f); }
  for (int i = tid; i < 64 * 32; i += 128) { int c = i >> 5, f = i & 31;
    sW2T[c * 32 + f] = (_Float16)W2[f * 64 + c]; }
  if (tid < 32) sB1[tid] = b1[tid];
  if (tid < 96) sTE[tid] = temb[tid];
  __syncthreads();

  int wave = tid >> 5, lane = tid & 31;
  int group = blockIdx.x * WPB_F + wave;
  int e = group * 32 + lane;
  int src = ei[e], dst = ei[NEDGES + e];

  float ax = (pos[dst*3+0] - pos[src*3+0]) * NM2A_F;
  float ay = (pos[dst*3+1] - pos[src*3+1]) * NM2A_F;
  float az = (pos[dst*3+2] - pos[src*3+2]) * NM2A_F;
  float r  = sqrtf(ax*ax + ay*ay + az*az + 1e-12f);
  float inv_r = 1.0f / r;
  float rx = ax * inv_r, ry = ay * inv_r, rz = az * inv_r;
  float x = r * (1.0f / RCUT_F);
  float x2 = x*x, x3 = x2*x, x6 = x3*x3, x7 = x6*x, x8 = x7*x;
  float env = 1.0f - 28.0f*x6 + 48.0f*x7 - 21.0f*x8;
  env = (x < 1.0f) ? env : 0.0f;

  _Float16* bro = &sBas[wave][lane * 32];
#pragma unroll
  for (int n = 0; n < NBAS; ++n) bro[n] = (_Float16)(__sinf(PI_F * (float)(n + 1) * x) * inv_r);
#pragma unroll
  for (int k = NBAS; k < 32; ++k) bro[k] = (_Float16)0.0f;
  __builtin_amdgcn_wave_barrier();

  int c16 = lane & 15, mb = (lane & 16) ? 8 : 0;
  // MLP1: basis(32x32pad) @ W1(32x32) + b1, silu -> sAct (f16)
#pragma unroll
  for (int m = 0; m < 2; ++m) {
    v16h A = ldsA(sBas[wave], 32, m, 0, lane);
#pragma unroll
    for (int nt = 0; nt < 2; ++nt) {
      v16h B = ldsB(sW1T, 32, nt, 0, lane);
      float bb = sB1[nt * 16 + c16];
      v8f C = {bb, bb, bb, bb, bb, bb, bb, bb};
      v8f D = wmma_f16(A, B, C);
#pragma unroll
      for (int i = 0; i < 8; ++i) {
        float t = D[i];
        sAct[wave][(m * 16 + mb + i) * 32 + nt * 16 + c16] = (_Float16)(t * sigf(t));
      }
    }
  }
  __builtin_amdgcn_wave_barrier();
  // MLP2: act(32x32) @ W2(32x64) -> sOut (f32, raw, env applied later)
#pragma unroll
  for (int m = 0; m < 2; ++m) {
    v16h A = ldsA(sAct[wave], 32, m, 0, lane);
#pragma unroll
    for (int nt = 0; nt < 4; ++nt) {
      v16h B = ldsB(sW2T, 32, nt, 0, lane);
      v8f Z = {0, 0, 0, 0, 0, 0, 0, 0};
      v8f D = wmma_f16(A, B, Z);
#pragma unroll
      for (int i = 0; i < 8; ++i) sOut[wave][(m * 16 + mb + i) * 64 + nt * 16 + c16] = D[i];
    }
  }
  __builtin_amdgcn_wave_barrier();

  // scatter: s[dst] += hs*w0 ; v[dst] += hs*w1*rhat
  const float* orow = &sOut[wave][lane * 64];
  const float* hrow = &sTE[types[src] * 32];
  float* srow = Sacc + (size_t)dst * 32;
  float* vrow = Vacc + (size_t)dst * 96;
#pragma unroll 4
  for (int f = 0; f < 32; ++f) {
    float hw = hrow[f];
    float w0 = orow[f] * env, w1 = orow[32 + f] * env;
    atomicAdd(&srow[f], hw * w0);
    float c = hw * w1;
    atomicAdd(&vrow[f * 3 + 0], c * rx);
    atomicAdd(&vrow[f * 3 + 1], c * ry);
    atomicAdd(&vrow[f * 3 + 2], c * rz);
  }
}

// ------------------------------------------------------------------ atom head fwd+bwd
__global__ __launch_bounds__(256) void atom_kernel(
    const float* __restrict__ Sacc, const float* __restrict__ Vacc,
    const float* __restrict__ Wself, const float* __restrict__ Wo1,
    const float* __restrict__ bo1, const float* __restrict__ Wo2,
    const int* __restrict__ types, const float* __restrict__ temb,
    float* __restrict__ dS, float* __restrict__ dV, float* __restrict__ scal)
{
  __shared__ float sWs[1024], sW1[2048], sW2[32], sB[32], sTE[96];
  __shared__ float red[256];
  int tid = threadIdx.x;
  for (int i = tid; i < 1024; i += 256) sWs[i] = Wself[i];
  for (int i = tid; i < 2048; i += 256) sW1[i] = Wo1[i];
  if (tid < 32) { sW2[tid] = Wo2[tid]; sB[tid] = bo1[tid]; }
  if (tid < 96) sTE[tid] = temb[tid];
  __syncthreads();

  int a = blockIdx.x * 256 + tid;
  float pa = 0.0f;
  if (a < NATOMS) {
    int ty = types[a];
    float u[32];
#pragma unroll
    for (int j = 0; j < 32; ++j) u[j] = sTE[ty * 32 + j];
    for (int i = 0; i < 32; ++i) {
      float si = Sacc[(size_t)a * 32 + i];
#pragma unroll
      for (int j = 0; j < 32; ++j) u[j] += si * sWs[i * 32 + j];
    }
    float vn[32];
#pragma unroll
    for (int f = 0; f < 32; ++f) {
      const float* vp = &Vacc[(size_t)a * 96 + f * 3];
      vn[f] = sqrtf(vp[0]*vp[0] + vp[1]*vp[1] + vp[2]*vp[2] + 1e-12f);
    }
    float z[32];
#pragma unroll
    for (int j = 0; j < 32; ++j) z[j] = sB[j];
    for (int i = 0; i < 32; ++i) {
      float ui = u[i];
#pragma unroll
      for (int j = 0; j < 32; ++j) z[j] += ui * sW1[i * 32 + j];
    }
    for (int f = 0; f < 32; ++f) {
      float vf = vn[f];
#pragma unroll
      for (int j = 0; j < 32; ++j) z[j] += vf * sW1[(32 + f) * 32 + j];
    }
    float dz[32];
#pragma unroll
    for (int j = 0; j < 32; ++j) {
      float sg = sigf(z[j]);
      pa += z[j] * sg * sW2[j];
      dz[j] = sW2[j] * sg * (1.0f + z[j] * (1.0f - sg));
    }
    // du -> reuse u ; dvnorm -> reuse z
    for (int i = 0; i < 32; ++i) {
      float acc = 0.0f;
#pragma unroll
      for (int j = 0; j < 32; ++j) acc += dz[j] * sW1[i * 32 + j];
      u[i] = acc;
    }
    for (int f = 0; f < 32; ++f) {
      float acc = 0.0f;
#pragma unroll
      for (int j = 0; j < 32; ++j) acc += dz[j] * sW1[(32 + f) * 32 + j];
      z[f] = acc;
    }
    for (int i = 0; i < 32; ++i) {
      float acc = 0.0f;
#pragma unroll
      for (int j = 0; j < 32; ++j) acc += u[j] * sWs[i * 32 + j];
      dS[(size_t)a * 32 + i] = acc;
    }
#pragma unroll
    for (int f = 0; f < 32; ++f) {
      const float* vp = &Vacc[(size_t)a * 96 + f * 3];
      float sc = z[f] / vn[f];
      float* dvp = &dV[(size_t)a * 96 + f * 3];
      dvp[0] = sc * vp[0]; dvp[1] = sc * vp[1]; dvp[2] = sc * vp[2];
    }
  }
  red[tid] = pa; __syncthreads();
  for (int o = 128; o; o >>= 1) { if (tid < o) red[tid] += red[tid + o]; __syncthreads(); }
  if (tid == 0) atomicAdd(&scal[0], red[0]);
}

// ------------------------------------------------------------------ edge backward
#define WPB_B 2
__global__ __launch_bounds__(64) void edge_bwd_kernel(
    const float* __restrict__ pos, const float* __restrict__ W1,
    const float* __restrict__ b1, const float* __restrict__ W2,
    const int* __restrict__ types, const int* __restrict__ ei,
    const float* __restrict__ temb, const float* __restrict__ dS,
    const float* __restrict__ dV, float* __restrict__ Gacc)
{
  __shared__ _Float16 sBas[WPB_B][32 * 32];
  __shared__ _Float16 sAct[WPB_B][32 * 32];
  __shared__ float    sT  [WPB_B][32 * 32];
  __shared__ float    sOut[WPB_B][32 * 64];
  __shared__ _Float16 sDW [WPB_B][32 * 64];
  __shared__ _Float16 sDT [WPB_B][32 * 32];
  __shared__ float    sDB [WPB_B][32 * 8];
  __shared__ _Float16 sW1T[32 * 32];   // [n][k] = W1[k][n], padded
  __shared__ _Float16 sW2T[64 * 32];   // [c][f] = W2[f][c]
  __shared__ _Float16 sW2R[32 * 64];   // [f][c] = W2 row-major (B for @W2^T)
  __shared__ _Float16 sW1R[16 * 32];   // [n][f] = W1[n][f], rows 8..15 zero (B for @W1^T)
  __shared__ float    sB1[32];
  __shared__ float    sTE[96];

  int tid = threadIdx.x;
  for (int i = tid; i < 32 * 32; i += 64) { int n = i >> 5, k = i & 31;
    sW1T[n * 32 + k] = (_Float16)((k < NBAS) ? W1[k * FDIM + n] : 0.0f); }
  for (int i = tid; i < 64 * 32; i += 64) { int c = i >> 5, f = i & 31;
    sW2T[c * 32 + f] = (_Float16)W2[f * 64 + c]; }
  for (int i = tid; i < 32 * 64; i += 64) sW2R[i] = (_Float16)W2[i];
  for (int i = tid; i < 16 * 32; i += 64) { int n = i >> 5, f = i & 31;
    sW1R[i] = (_Float16)((n < NBAS) ? W1[n * FDIM + f] : 0.0f); }
  if (tid < 32) sB1[tid] = b1[tid];
  for (int i = tid; i < 96; i += 64) sTE[i] = temb[i];
  __syncthreads();

  int wave = tid >> 5, lane = tid & 31;
  int group = blockIdx.x * WPB_B + wave;
  int e = group * 32 + lane;
  int src = ei[e], dst = ei[NEDGES + e];

  float ax = (pos[dst*3+0] - pos[src*3+0]) * NM2A_F;
  float ay = (pos[dst*3+1] - pos[src*3+1]) * NM2A_F;
  float az = (pos[dst*3+2] - pos[src*3+2]) * NM2A_F;
  float r  = sqrtf(ax*ax + ay*ay + az*az + 1e-12f);
  float inv_r = 1.0f / r;
  float rx = ax * inv_r, ry = ay * inv_r, rz = az * inv_r;
  float x = r * (1.0f / RCUT_F);
  float x2 = x*x, x3 = x2*x, x5 = x2*x3, x6 = x3*x3, x7 = x6*x, x8 = x7*x;
  float env  = 1.0f - 28.0f*x6 + 48.0f*x7 - 21.0f*x8;
  float envp = -168.0f*x5 + 336.0f*x6 - 168.0f*x7;   // d env / d x
  bool inside = (x < 1.0f);
  env  = inside ? env  : 0.0f;
  envp = inside ? envp : 0.0f;

  _Float16* bro = &sBas[wave][lane * 32];
#pragma unroll
  for (int n = 0; n < NBAS; ++n) bro[n] = (_Float16)(__sinf(PI_F * (float)(n + 1) * x) * inv_r);
#pragma unroll
  for (int k = NBAS; k < 32; ++k) bro[k] = (_Float16)0.0f;
  __builtin_amdgcn_wave_barrier();

  int c16 = lane & 15, mb = (lane & 16) ? 8 : 0;
  // recompute MLP1, keep pre-activation t and a=silu(t)
#pragma unroll
  for (int m = 0; m < 2; ++m) {
    v16h A = ldsA(sBas[wave], 32, m, 0, lane);
#pragma unroll
    for (int nt = 0; nt < 2; ++nt) {
      v16h B = ldsB(sW1T, 32, nt, 0, lane);
      float bb = sB1[nt * 16 + c16];
      v8f C = {bb, bb, bb, bb, bb, bb, bb, bb};
      v8f D = wmma_f16(A, B, C);
#pragma unroll
      for (int i = 0; i < 8; ++i) {
        int idx = (m * 16 + mb + i) * 32 + nt * 16 + c16;
        float t = D[i];
        sT[wave][idx] = t;
        sAct[wave][idx] = (_Float16)(t * sigf(t));
      }
    }
  }
  __builtin_amdgcn_wave_barrier();
  // recompute MLP2 raw output
#pragma unroll
  for (int m = 0; m < 2; ++m) {
    v16h A = ldsA(sAct[wave], 32, m, 0, lane);
#pragma unroll
    for (int nt = 0; nt < 4; ++nt) {
      v16h B = ldsB(sW2T, 32, nt, 0, lane);
      v8f Z = {0, 0, 0, 0, 0, 0, 0, 0};
      v8f D = wmma_f16(A, B, Z);
#pragma unroll
      for (int i = 0; i < 8; ++i) sOut[wave][(m * 16 + mb + i) * 64 + nt * 16 + c16] = D[i];
    }
  }
  __builtin_amdgcn_wave_barrier();

  // per-edge: gather ds[dst], dv[dst]; build d(rw_raw), denv, drhat
  float denv = 0.0f, dhx = 0.0f, dhy = 0.0f, dhz = 0.0f;
  {
    const float* orow = &sOut[wave][lane * 64];
    const float* hrow = &sTE[types[src] * 32];
    const float* dsr  = dS + (size_t)dst * 32;
    const float* dvr  = dV + (size_t)dst * 96;
    _Float16* dwr = &sDW[wave][lane * 64];
#pragma unroll 4
    for (int f = 0; f < 32; ++f) {
      float rr0 = orow[f], rr1 = orow[32 + f];
      float hw  = hrow[f];
      float dw0 = hw * dsr[f];
      float d0 = dvr[f*3+0], d1 = dvr[f*3+1], d2 = dvr[f*3+2];
      float dot = d0 * rx + d1 * ry + d2 * rz;
      float dw1 = hw * dot;
      float hw1 = hw * (rr1 * env);
      dhx += d0 * hw1; dhy += d1 * hw1; dhz += d2 * hw1;
      denv += dw0 * rr0 + dw1 * rr1;
      dwr[f]      = (_Float16)(dw0 * env);
      dwr[32 + f] = (_Float16)(dw1 * env);
    }
  }
  __builtin_amdgcn_wave_barrier();

  // da = dRW @ W2^T  (K=64), then dt = da * silu'(t)
#pragma unroll
  for (int m = 0; m < 2; ++m) {
    v16h A0 = ldsA(sDW[wave], 64, m, 0, lane);
    v16h A1 = ldsA(sDW[wave], 64, m, 32, lane);
#pragma unroll
    for (int nt = 0; nt < 2; ++nt) {
      v16h B0 = ldsB(sW2R, 64, nt, 0, lane);
      v16h B1 = ldsB(sW2R, 64, nt, 32, lane);
      v8f Z = {0, 0, 0, 0, 0, 0, 0, 0};
      v8f D = wmma_f16(A0, B0, Z);
      D = wmma_f16(A1, B1, D);
#pragma unroll
      for (int i = 0; i < 8; ++i) {
        int idx = (m * 16 + mb + i) * 32 + nt * 16 + c16;
        float t = sT[wave][idx];
        float sg = sigf(t);
        float dsl = sg * (1.0f + t * (1.0f - sg));
        sDT[wave][idx] = (_Float16)(D[i] * dsl);
      }
    }
  }
  __builtin_amdgcn_wave_barrier();

  // dbasis = dt @ W1^T  (N=8 valid of 16)
#pragma unroll
  for (int m = 0; m < 2; ++m) {
    v16h A = ldsA(sDT[wave], 32, m, 0, lane);
    v16h B = ldsB(sW1R, 32, 0, 0, lane);
    v8f Z = {0, 0, 0, 0, 0, 0, 0, 0};
    v8f D = wmma_f16(A, B, Z);
    if (c16 < 8) {
#pragma unroll
      for (int i = 0; i < 8; ++i) sDB[wave][(m * 16 + mb + i) * 8 + c16] = D[i];
    }
  }
  __builtin_amdgcn_wave_barrier();

  // chain to positions
  float dr = denv * envp * (1.0f / RCUT_F);
  const float* dbr = &sDB[wave][lane * 8];
#pragma unroll
  for (int n = 0; n < NBAS; ++n) {
    float arg = PI_F * (float)(n + 1) * x;
    float sn = __sinf(arg), cs = __cosf(arg);
    dr += dbr[n] * ((PI_F * (float)(n + 1) * (1.0f / RCUT_F)) * cs * inv_r - sn * inv_r * inv_r);
  }
  float ddot = rx * dhx + ry * dhy + rz * dhz;
  float gx = (dhx - rx * ddot) * inv_r + dr * rx;
  float gy = (dhy - ry * ddot) * inv_r + dr * ry;
  float gz = (dhz - rz * ddot) * inv_r + dr * rz;
  atomicAdd(&Gacc[(size_t)dst * 3 + 0],  gx);
  atomicAdd(&Gacc[(size_t)dst * 3 + 1],  gy);
  atomicAdd(&Gacc[(size_t)dst * 3 + 2],  gz);
  atomicAdd(&Gacc[(size_t)src * 3 + 0], -gx);
  atomicAdd(&Gacc[(size_t)src * 3 + 1], -gy);
  atomicAdd(&Gacc[(size_t)src * 3 + 2], -gz);
}

// ------------------------------------------------------------------ net-force + mass reduce
__global__ __launch_bounds__(256) void net_kernel(
    const float* __restrict__ G, const float* __restrict__ masses, float* __restrict__ scal)
{
  __shared__ float red[256];
  int tid = threadIdx.x;
  int a = blockIdx.x * 256 + tid;
  float q[4] = {0.0f, 0.0f, 0.0f, 0.0f};
  if (a < NATOMS) {
    q[0] = -G[a*3+0] * FSCALE_F;
    q[1] = -G[a*3+1] * FSCALE_F;
    q[2] = -G[a*3+2] * FSCALE_F;
    q[3] = masses[a];
  }
#pragma unroll
  for (int c = 0; c < 4; ++c) {
    red[tid] = q[c]; __syncthreads();
    for (int o = 128; o; o >>= 1) { if (tid < o) red[tid] += red[tid + o]; __syncthreads(); }
    if (tid == 0) atomicAdd(&scal[1 + c], red[0]);
    __syncthreads();
  }
}

// ------------------------------------------------------------------ finalize
__global__ __launch_bounds__(256) void finalize_kernel(
    const float* __restrict__ G, const float* __restrict__ masses,
    const float* __restrict__ scal, float* __restrict__ out)
{
  int a = blockIdx.x * 256 + threadIdx.x;
  if (a == 0) out[0] = scal[0] * EV2KJ_F;
  if (a < NATOMS) {
    float coef = masses[a] / scal[4];
#pragma unroll
    for (int k = 0; k < 3; ++k)
      out[1 + a * 3 + k] = -G[a * 3 + k] * FSCALE_F - coef * scal[1 + k];
  }
}

// ------------------------------------------------------------------ host
extern "C" void kernel_launch(void* const* d_in, const int* in_sizes, int n_in,
                              void* d_out, int out_size, void* d_ws, size_t ws_size,
                              hipStream_t stream) {
  (void)in_sizes; (void)n_in; (void)out_size; (void)ws_size;
  const float* positions = (const float*)d_in[0];
  const float* masses    = (const float*)d_in[1];
  const float* temb      = (const float*)d_in[2];
  const float* W1        = (const float*)d_in[3];
  const float* b1        = (const float*)d_in[4];
  const float* W2        = (const float*)d_in[5];
  const float* Wself     = (const float*)d_in[6];
  const float* Wo1       = (const float*)d_in[7];
  const float* bo1       = (const float*)d_in[8];
  const float* Wo2       = (const float*)d_in[9];
  const int*   types     = (const int*)d_in[10];
  const int*   ei        = (const int*)d_in[11];
  float* out = (float*)d_out;
  float* W   = (float*)d_ws;
  float* S   = W + OFF_S;
  float* V   = W + OFF_V;
  float* dSw = W + OFF_DS;
  float* dVw = W + OFF_DV;
  float* G   = W + OFF_G;
  float* sc  = W + OFF_SC;

  zero_ws_kernel<<<(WS_FLOATS + 255) / 256, 256, 0, stream>>>(W, WS_FLOATS);
  edge_fwd_kernel<<<NEDGES / (32 * WPB_F), 128, 0, stream>>>(
      positions, W1, b1, W2, types, ei, temb, S, V);
  atom_kernel<<<(NATOMS + 255) / 256, 256, 0, stream>>>(
      S, V, Wself, Wo1, bo1, Wo2, types, temb, dSw, dVw, sc);
  edge_bwd_kernel<<<NEDGES / (32 * WPB_B), 64, 0, stream>>>(
      positions, W1, b1, W2, types, ei, temb, dSw, dVw, G);
  net_kernel<<<(NATOMS + 255) / 256, 256, 0, stream>>>(G, masses, sc);
  finalize_kernel<<<(NATOMS + 255) / 256, 256, 0, stream>>>(G, masses, sc, out);
}
